// ScaleAttention_89824946029319
// MI455X (gfx1250) — matrix operand order repaired and checked by reading, not compiled, not run on hardware
//
#include <hip/hip_runtime.h>
#include <stdint.h>

// CDNA5 (gfx1250) flash-attention:  O = softmax(mask(Q K^T * 0.125)) V
// B=2, H=16, S=2048, D=64, fp32 in/out, bool mask (True = masked).
// GEMMs on v_wmma_f32_16x16x32_f16 (wave32); K/V tiles double-buffered in LDS
// via global_load_async_to_lds (ASYNCcnt) — builtins confirmed present.

typedef __attribute__((ext_vector_type(16))) _Float16 v16h;
typedef __attribute__((ext_vector_type(8)))  float    v8f;
typedef uint32_t __attribute__((may_alias))  u32a;
typedef float f4a __attribute__((ext_vector_type(4), may_alias));
typedef int   v4i __attribute__((vector_size(16)));   // matches builtin param type

#ifndef __has_builtin
#define __has_builtin(x) 0
#endif
#if __has_builtin(__builtin_amdgcn_global_load_async_to_lds_b128) && \
    __has_builtin(__builtin_amdgcn_global_load_async_to_lds_b32)
#define HAS_ASYNC_LDS 1
#else
#define HAS_ASYNC_LDS 0
#endif

#define AS1 __attribute__((address_space(1)))
#define AS3 __attribute__((address_space(3)))

#define S_LEN      2048
#define D_HEAD     64
#define BH_TOTAL   32           // B*H
#define TILE_KEYS  32
#define WAVES_WG   8
#define ROWS_WAVE  16
#define ROWS_WG    (WAVES_WG * ROWS_WAVE)   // 128 query rows / workgroup
#define K_STR      68           // f32 units: 68 % 64 banks = 4 -> conflict-free b128 reads
#define VT_STR     36           // f32 units: 36n mod 64 distinct for n=0..15
#define P_STRIDE   40           // f16 units

union Frag16 { v16h h; uint32_t u[8]; };

#define WMMA_F16(A, B, C) \
    __builtin_amdgcn_wmma_f32_16x16x32_f16(false, (A).h, false, (B).h, (short)0, (C), false, false)

__device__ __forceinline__ uint32_t pack2h(float a, float b) {
    union { _Float16 h[2]; uint32_t u; } p;
    p.h[0] = (_Float16)a; p.h[1] = (_Float16)b;
    return p.u;
}

// packed f32x2 -> f16x2 (v_cvt_pk_rtz_f16_f32)
__device__ __forceinline__ uint32_t cvt2(float a, float b) {
    return __builtin_bit_cast(uint32_t, __builtin_amdgcn_cvt_pkrtz(a, b));
}

__device__ __forceinline__ void async_b128(void* lds, const void* g) {
#if HAS_ASYNC_LDS
    __builtin_amdgcn_global_load_async_to_lds_b128((AS1 v4i*)g, (AS3 v4i*)lds, 0, 0);
#else
    *(f4a*)lds = *(const f4a*)g;          // sync fallback
#endif
}
__device__ __forceinline__ void async_b32(void* lds, const void* g) {
#if HAS_ASYNC_LDS
    __builtin_amdgcn_global_load_async_to_lds_b32((AS1 int*)g, (AS3 int*)lds, 0, 0);
#else
    *(float*)lds = *(const float*)g;      // sync fallback
#endif
}
__device__ __forceinline__ void wait_async0() {
#if HAS_ASYNC_LDS
#if __has_builtin(__builtin_amdgcn_s_wait_asynccnt)
    __builtin_amdgcn_s_wait_asynccnt(0);
#else
    asm volatile("s_wait_asynccnt 0" ::: "memory");
#endif
#endif
}

// Stage one 32-key tile: K -> [key][d] f32 (padded), V -> [d][key] f32 (transposed scatter).
__device__ __forceinline__ void issue_tile_copies(const float* __restrict__ Kb,
                                                  const float* __restrict__ Vb,
                                                  int kb, float* kdst, float* vdst, int tid) {
#pragma unroll
    for (int it = 0; it < 2; ++it) {              // 32x64 f32 = 512 float4, 256 thr
        const int f4idx = it * 256 + tid;
        const int key   = f4idx >> 4;
        const int d4    = (f4idx & 15) << 2;
        async_b128(&kdst[key * K_STR + d4], Kb + (size_t)(kb + key) * D_HEAD + d4);
    }
#pragma unroll
    for (int it = 0; it < 8; ++it) {              // per-element scatter = free transpose
        const int idx = it * 256 + tid;
        const int key = idx >> 6;
        const int d   = idx & 63;
        async_b32(&vdst[d * VT_STR + key], Vb + (size_t)(kb + key) * D_HEAD + d);
    }
}

// Build a 16-bit 32x16 B-fragment from 16 consecutive f32 in LDS (row K-pairs per VGPR).
__device__ __forceinline__ void load_bfrag(Frag16& f, const float* base) {
    f4a x0 = *(const f4a*)(base);
    f4a x1 = *(const f4a*)(base + 4);
    f4a x2 = *(const f4a*)(base + 8);
    f4a x3 = *(const f4a*)(base + 12);
    f.u[0] = cvt2(x0.x, x0.y); f.u[1] = cvt2(x0.z, x0.w);
    f.u[2] = cvt2(x1.x, x1.y); f.u[3] = cvt2(x1.z, x1.w);
    f.u[4] = cvt2(x2.x, x2.y); f.u[5] = cvt2(x2.z, x2.w);
    f.u[6] = cvt2(x3.x, x3.y); f.u[7] = cvt2(x3.z, x3.w);
}

__global__ __launch_bounds__(256)
void flash_attn_wmma_f16(const float* __restrict__ Q,
                         const float* __restrict__ K,
                         const float* __restrict__ V,
                         const unsigned char* __restrict__ mask,
                         float* __restrict__ O)
{
    __shared__ __align__(16) float    kbuf[2][TILE_KEYS * K_STR];   // [key][d] f32
    __shared__ __align__(16) float    vbuf[2][D_HEAD * VT_STR];     // [d][key] f32 (transposed)
    __shared__ __align__(16) _Float16 ptile[WAVES_WG * ROWS_WAVE * P_STRIDE];

    const int tid  = threadIdx.x;
    const int wave = tid >> 5;
    const int lane = tid & 31;
    const int half = lane >> 4;      // 16-lane half of the wave
    const int n    = lane & 15;      // B/C fragment column

    const int bh = blockIdx.y;
    const int q0 = blockIdx.x * ROWS_WG + wave * ROWS_WAVE;

    const float scale  = 0.125f;                 // 1/sqrt(64)
    const float NEGINF = -__builtin_inff();

    const float* Qb = Q + (size_t)bh * S_LEN * D_HEAD;
    const float* Kb = K + (size_t)bh * S_LEN * D_HEAD;
    const float* Vb = V + (size_t)bh * S_LEN * D_HEAD;
    float*       Ob = O + (size_t)bh * S_LEN * D_HEAD;

    // ---- Q A-fragments (16x32 f16, two chunks over D=64), loaded once ----
    // A layout: lane m = lane%16; VGPR r holds K pair half*16 + r*2.
    Frag16 aq0, aq1;
#pragma unroll
    for (int r = 0; r < 8; ++r) {
        const int dbase = half * 16 + r * 2;
        const float* qp = Qb + (size_t)(q0 + n) * D_HEAD + dbase;
        aq0.u[r] = pack2h(qp[0],  qp[1]);
        aq1.u[r] = pack2h(qp[32], qp[33]);
    }

    float row_max[8], row_sum[8];
    v8f   of[4] = {v8f{}, v8f{}, v8f{}, v8f{}};  // O accum: of[t][j] = (m=j+8h, d=16t+n)
#pragma unroll
    for (int j = 0; j < 8; ++j) { row_max[j] = NEGINF; row_sum[j] = 0.0f; }

    _Float16* pbase = &ptile[wave * ROWS_WAVE * P_STRIDE];

    // ---- prologue: start async fill of buffer 0 ----
    issue_tile_copies(Kb, Vb, 0, kbuf[0], vbuf[0], tid);

    int buf = 0;
    for (int kb = 0; kb < S_LEN; kb += TILE_KEYS, buf ^= 1) {
        wait_async0();          // my async copies for this tile done
        __syncthreads();        // everyone's copies done; prev reads of other buf done

        if (kb + TILE_KEYS < S_LEN)   // overlap next tile fill with this tile's compute
            issue_tile_copies(Kb, Vb, kb + TILE_KEYS, kbuf[buf ^ 1], vbuf[buf ^ 1], tid);

        const float* kcur = kbuf[buf];
        const float* vcur = vbuf[buf];

        // ---- scores: S(16x32) = Q(16x64) x K^T(64x32), two 16-key groups ----
        // B layout: col N = lane%16; VGPR r holds K rows half*16 + 2r, 2r+1.
        v8f c0 = {}, c1 = {};
        {
            Frag16 b;
            load_bfrag(b, &kcur[n * K_STR + half * 16]);              // g0, d[0,32)
            c0 = WMMA_F16(aq0, b, c0);
            load_bfrag(b, &kcur[n * K_STR + 32 + half * 16]);         // g0, d[32,64)
            c0 = WMMA_F16(aq1, b, c0);
            load_bfrag(b, &kcur[(16 + n) * K_STR + half * 16]);       // g1, d[0,32)
            c1 = WMMA_F16(aq0, b, c1);
            load_bfrag(b, &kcur[(16 + n) * K_STR + 32 + half * 16]);  // g1, d[32,64)
            c1 = WMMA_F16(aq1, b, c1);
        }

        // ---- mask + scale + online softmax (rows m = j + 8*half) ----
        float factor[8], p0[8], p1[8];
#pragma unroll
        for (int j = 0; j < 8; ++j) {
            const int m = j + 8 * half;
            const unsigned char* mp = mask + (size_t)(q0 + m) * S_LEN + kb;
            float s0 = mp[n]      ? NEGINF : c0[j] * scale;
            float s1 = mp[16 + n] ? NEGINF : c1[j] * scale;
            float mb = fmaxf(s0, s1);
#pragma unroll
            for (int off = 1; off < 16; off <<= 1)       // xor<16 stays inside the half
                mb = fmaxf(mb, __shfl_xor(mb, off));
            const float nm = fmaxf(row_max[j], mb);
            const float f  = (nm == NEGINF) ? 1.0f : __expf(row_max[j] - nm);
            row_max[j] = nm;
            factor[j]  = f;
            p0[j] = (nm == NEGINF) ? 0.0f : __expf(s0 - nm);
            p1[j] = (nm == NEGINF) ? 0.0f : __expf(s1 - nm);
            row_sum[j] = row_sum[j] * f + p0[j] + p1[j]; // lane-partial; reduced at end
        }

        // ---- rescale O accumulators, stage P (f16) in per-wave LDS scratch ----
#pragma unroll
        for (int j = 0; j < 8; ++j) {
            of[0][j] *= factor[j]; of[1][j] *= factor[j];
            of[2][j] *= factor[j]; of[3][j] *= factor[j];
            const int m = j + 8 * half;
            pbase[m * P_STRIDE + n]      = (_Float16)p0[j];
            pbase[m * P_STRIDE + 16 + n] = (_Float16)p1[j];
        }
        // same-wave LDS ops are in-order: no barrier needed before re-reading P

        // ---- O += P(16x32) x V(32x64) ----
        Frag16 pa;
#pragma unroll
        for (int r = 0; r < 8; ++r) {
            const int kpair = half * 16 + r * 2;
            pa.u[r] = *(const u32a*)&pbase[n * P_STRIDE + kpair];
        }
#pragma unroll
        for (int t = 0; t < 4; ++t) {
            Frag16 bv;
            load_bfrag(bv, &vcur[(t * 16 + n) * VT_STR + half * 16]);
            of[t] = WMMA_F16(pa, bv, of[t]);
        }
    }

    // ---- epilogue: reduce row sums across the 16-lane half, normalize, store ----
#pragma unroll
    for (int j = 0; j < 8; ++j) {
        float s = row_sum[j];
#pragma unroll
        for (int off = 1; off < 16; off <<= 1)
            s += __shfl_xor(s, off);
        const float inv = (s > 0.0f) ? (1.0f / s) : 0.0f;   // fully-masked rows -> 0
        const int m = j + 8 * half;
        float* op = Ob + (size_t)(q0 + m) * D_HEAD + n;
        op[0]  = of[0][j] * inv;
        op[16] = of[1][j] * inv;
        op[32] = of[2][j] * inv;
        op[48] = of[3][j] * inv;
    }
}

extern "C" void kernel_launch(void* const* d_in, const int* in_sizes, int n_in,
                              void* d_out, int out_size, void* d_ws, size_t ws_size,
                              hipStream_t stream) {
    (void)in_sizes; (void)n_in; (void)out_size; (void)d_ws; (void)ws_size;
    const float*         Q    = (const float*)d_in[0];
    const float*         K    = (const float*)d_in[1];
    const float*         V    = (const float*)d_in[2];
    const unsigned char* mask = (const unsigned char*)d_in[3];  // jax bool = 1 byte
    float*               out  = (float*)d_out;

    dim3 grid(S_LEN / ROWS_WG, BH_TOTAL);   // (16, 32)
    dim3 block(256);                        // 8 wave32s
    flash_attn_wmma_f16<<<grid, block, 0, stream>>>(Q, K, V, mask, out);
}
